// HierarchicalContrastiveLoss_84645215470200
// MI455X (gfx1250) — matrix operand order repaired and checked by reading, not compiled, not run on hardware
//
#include <hip/hip_runtime.h>
#include <hip/hip_bf16.h>

typedef _Float16 v16h __attribute__((ext_vector_type(16)));
typedef _Float16 v8h  __attribute__((ext_vector_type(8)));
typedef _Float16 v4h  __attribute__((ext_vector_type(4)));
typedef float    v8f  __attribute__((ext_vector_type(8)));

#define INV_TAU 14.285714285714286f      /* 1/0.07 */
#define K_EXP2  20.60992915555662f       /* (1/tau) * log2(e) */
#define EPSN    1e-12f
#define MROWS   64                       /* rows per strip (register-resident A) */
#define CS      32                       /* column chunks per strip */

// ---------------------------------------------------------------------------
// Kernel 1: L2-normalize rows, emit f16 concat Z, pos/tau, selfexp (diagonal
// correction term), zero rowsum & out. One wave per row pair.
// ---------------------------------------------------------------------------
__global__ __launch_bounds__(256) void ntxent_norm(
    const float* __restrict__ zi, const float* __restrict__ zj,
    _Float16* __restrict__ z16, float* __restrict__ pos,
    float* __restrict__ selfexp, float* __restrict__ rowsum,
    float* __restrict__ out, int N)
{
    int gid = blockIdx.x * blockDim.x + threadIdx.x;
    if (gid < 2 * N) rowsum[gid] = 0.0f;
    if (gid == 0)    out[0] = 0.0f;

    int r    = blockIdx.x * 8 + (threadIdx.x >> 5);
    int lane = threadIdx.x & 31;
    if (r >= N) return;

    float4 vi = reinterpret_cast<const float4*>(zi + (size_t)r * 128)[lane];
    float4 vj = reinterpret_cast<const float4*>(zj + (size_t)r * 128)[lane];

    float ssi = vi.x*vi.x + vi.y*vi.y + vi.z*vi.z + vi.w*vi.w;
    float ssj = vj.x*vj.x + vj.y*vj.y + vj.z*vj.z + vj.w*vj.w;
    #pragma unroll
    for (int off = 16; off; off >>= 1) {
        ssi += __shfl_xor(ssi, off, 32);
        ssj += __shfl_xor(ssj, off, 32);
    }
    float si = 1.0f / fmaxf(sqrtf(ssi), EPSN);
    float sj = 1.0f / fmaxf(sqrtf(ssj), EPSN);
    vi.x *= si; vi.y *= si; vi.z *= si; vi.w *= si;
    vj.x *= sj; vj.y *= sj; vj.z *= sj; vj.w *= sj;

    float pd = vi.x*vj.x + vi.y*vj.y + vi.z*vj.z + vi.w*vj.w;
    #pragma unroll
    for (int off = 16; off; off >>= 1) pd += __shfl_xor(pd, off, 32);
    if (lane == 0) pos[r] = pd * INV_TAU;

    v4h hi = { (_Float16)vi.x, (_Float16)vi.y, (_Float16)vi.z, (_Float16)vi.w };
    v4h hj = { (_Float16)vj.x, (_Float16)vj.y, (_Float16)vj.z, (_Float16)vj.w };
    *reinterpret_cast<v4h*>(z16 + (size_t)r * 128 + lane * 4)       = hi;
    *reinterpret_cast<v4h*>(z16 + (size_t)(N + r) * 128 + lane * 4) = hj;

    // Diagonal terms of exp((sim-1)/tau) computed from the f16-rounded rows;
    // kernel 3 subtracts them (no masking needed in the WMMA hot loop).
    float hix = (float)hi.x, hiy = (float)hi.y, hiz = (float)hi.z, hiw = (float)hi.w;
    float hjx = (float)hj.x, hjy = (float)hj.y, hjz = (float)hj.z, hjw = (float)hj.w;
    float di = hix*hix + hiy*hiy + hiz*hiz + hiw*hiw;
    float dj = hjx*hjx + hjy*hjy + hjz*hjz + hjw*hjw;
    #pragma unroll
    for (int off = 16; off; off >>= 1) {
        di += __shfl_xor(di, off, 32);
        dj += __shfl_xor(dj, off, 32);
    }
    if (lane == 0) {
        selfexp[r]     = __builtin_amdgcn_exp2f(di * K_EXP2 - K_EXP2);
        selfexp[N + r] = __builtin_amdgcn_exp2f(dj * K_EXP2 - K_EXP2);
    }
}

// ---------------------------------------------------------------------------
// Kernel 2: WMMA similarity + fixed-offset exp2-sum accumulation.
// Wave = (64-row strip, 1/CS of column tiles). Each 4KB B tile feeds 16 WMMAs
// (register-resident A: 16 x v16h = 128 VGPRs), balancing XDL vs load port.
// ---------------------------------------------------------------------------
__device__ __forceinline__ v16h load_frag(const _Float16* __restrict__ p,
                                          int o0, int o1)
{
    v8h lo = *reinterpret_cast<const v8h*>(p + o0);
    v8h hi = *reinterpret_cast<const v8h*>(p + o1);
    return __builtin_shufflevector(lo, hi, 0,1,2,3,4,5,6,7,8,9,10,11,12,13,14,15);
}

__global__ __launch_bounds__(256) void ntxent_sim(
    const _Float16* __restrict__ z16, float* __restrict__ rowsum,
    int tiles_per_chunk)
{
    int wid   = (blockIdx.x * blockDim.x + threadIdx.x) >> 5;
    int lane  = threadIdx.x & 31;
    int strip = wid >> 5;            // / CS
    int chunk = wid & (CS - 1);
    int m_base = strip * MROWS;
    int g    = lane >> 4;            // half-wave group
    int rmod = lane & 15;

    // Four A fragment sets: rows [m_base + 16t, m_base + 16t + 16), t=0..3.
    const _Float16* arow = z16 + (size_t)(m_base + rmod) * 128;
    v16h a[16];
    #pragma unroll
    for (int t = 0; t < 4; ++t) {
        const _Float16* ar = arow + (size_t)t * 16 * 128;
        #pragma unroll
        for (int kc = 0; kc < 4; ++kc)
            a[t * 4 + kc] = load_frag(ar, kc * 32 + 16 * g, kc * 32 + 8 + 16 * g);
    }

    float s[32];
    #pragma unroll
    for (int r = 0; r < 32; ++r) s[r] = 0.0f;

    // Running B pointer: 16 rows (one tile) per iteration.
    const _Float16* brow =
        z16 + (size_t)(chunk * tiles_per_chunk * 16 + rmod) * 128;

    for (int it = 0; it < tiles_per_chunk; ++it, brow += 16 * 128) {
        __builtin_prefetch(brow + 16 * 128, 0, 3);   // next tile, near cache

        // B fragments: B (32x16) in contiguous k order, same as A — K = 16*g + h, N = rmod.
        v16h b0 = load_frag(brow,  0 + 16 * g,   8 + 16 * g);
        v16h b1 = load_frag(brow, 32 + 16 * g,  40 + 16 * g);
        v16h b2 = load_frag(brow, 64 + 16 * g,  72 + 16 * g);
        v16h b3 = load_frag(brow, 96 + 16 * g, 104 + 16 * g);

        #pragma unroll
        for (int t = 0; t < 4; ++t) {
            v8f c = {};
            c = __builtin_amdgcn_wmma_f32_16x16x32_f16(false, a[t*4+0], false, b0, (short)0, c, false, false);
            c = __builtin_amdgcn_wmma_f32_16x16x32_f16(false, a[t*4+1], false, b1, (short)0, c, false, false);
            c = __builtin_amdgcn_wmma_f32_16x16x32_f16(false, a[t*4+2], false, b2, (short)0, c, false, false);
            c = __builtin_amdgcn_wmma_f32_16x16x32_f16(false, a[t*4+3], false, b3, (short)0, c, false, false);

            // exp((sim-1)/tau): one pk_fma + v_exp_f32 per element.
            #pragma unroll
            for (int r = 0; r < 8; ++r)
                s[t * 8 + r] += __builtin_amdgcn_exp2f(c[r] * K_EXP2 - K_EXP2);
        }
    }

    // Reduce across the 16 lanes that share row group g (columns differ).
    #pragma unroll
    for (int off = 1; off < 16; off <<= 1) {
        #pragma unroll
        for (int r = 0; r < 32; ++r) s[r] += __shfl_xor(s[r], off, 32);
    }
    if (rmod == 0) {
        #pragma unroll
        for (int t = 0; t < 4; ++t)
            #pragma unroll
            for (int r = 0; r < 8; ++r)
                atomicAdd(&rowsum[m_base + t * 16 + r + 8 * g], s[t * 8 + r]);
    }
}

// ---------------------------------------------------------------------------
// Kernel 3: loss = mean over 2N rows of
//           (log(rowsum - selfexp) + 1/tau - pos[row % N]).
// ---------------------------------------------------------------------------
__global__ __launch_bounds__(256) void ntxent_finish(
    const float* __restrict__ rowsum, const float* __restrict__ selfexp,
    const float* __restrict__ pos, float* __restrict__ out, int N)
{
    int row = blockIdx.x * blockDim.x + threadIdx.x;
    float v = 0.0f;
    if (row < 2 * N) {
        int pr = (row < N) ? row : row - N;
        v = __logf(rowsum[row] - selfexp[row]) + INV_TAU - pos[pr];
    }
    #pragma unroll
    for (int off = 16; off; off >>= 1) v += __shfl_xor(v, off, 32);
    if ((threadIdx.x & 31) == 0)
        atomicAdd(out, v * (1.0f / (2.0f * (float)N)));
}

// ---------------------------------------------------------------------------
extern "C" void kernel_launch(void* const* d_in, const int* in_sizes, int n_in,
                              void* d_out, int out_size, void* d_ws, size_t ws_size,
                              hipStream_t stream)
{
    const float* zi = (const float*)d_in[0];
    const float* zj = (const float*)d_in[1];
    float* out = (float*)d_out;

    int N = in_sizes[0] / 128;              // B*S = 4096 rows per view

    char* ws = (char*)d_ws;
    _Float16* z16 = (_Float16*)ws;                          // [2N][128] f16
    size_t z16_bytes = (size_t)2 * N * 128 * sizeof(_Float16);
    float* pos     = (float*)(ws + z16_bytes);              // [N]
    float* rowsum  = pos + N;                               // [2N]
    float* selfexp = rowsum + 2 * N;                        // [2N]

    // 1) normalize + f16 pack + pos + selfexp + zero-init
    ntxent_norm<<<N / 8, 256, 0, stream>>>(zi, zj, z16, pos, selfexp,
                                           rowsum, out, N);

    // 2) WMMA similarity: 2N/MROWS strips x CS column chunks, 1 wave each
    int col_tiles = (2 * N) / 16;               // 512
    int strips    = (2 * N) / MROWS;            // 128
    int tiles_per_chunk = col_tiles / CS;       // 16
    int waves = strips * CS;                    // 4096
    ntxent_sim<<<waves * 32 / 256, 256, 0, stream>>>(z16, rowsum,
                                                     tiles_per_chunk);

    // 3) final reduction to scalar loss
    ntxent_finish<<<(2 * N + 255) / 256, 256, 0, stream>>>(rowsum, selfexp,
                                                           pos, out, N);
}
